// LTNNetwork_59863254172230
// MI455X (gfx1250) — matrix operand order repaired and checked by reading, not compiled, not run on hardware
//
#include <hip/hip_runtime.h>
#include <hip/hip_bf16.h>
#include <math.h>
#include <stdint.h>

#define BB 128
#define NN 128
#define EE 32
#define FEAT 6

typedef __attribute__((ext_vector_type(16))) _Float16     v16h;
typedef __attribute__((ext_vector_type(8)))  float        v8f;
typedef __attribute__((ext_vector_type(4)))  unsigned int v4u;
typedef __attribute__((ext_vector_type(8)))  int          v8i;
typedef __attribute__((ext_vector_type(4)))  int          v4i;

__device__ __forceinline__ float sigmoidf_(float x) {
    return 1.0f / (1.0f + __expf(-x));
}

// A-frag element t -> K index (contiguous order k = t + 16*half, same as make_bfrag)
__device__ __forceinline__ int a_k(int t, int half) {
    return t + 16 * half;
}

__device__ __forceinline__ v8f wmma_f16(v16h a, v16h b) {
    v8f c = {};
    return __builtin_amdgcn_wmma_f32_16x16x32_f16(false, a, false, b, (short)0, c, false, false);
}

// Build B-frag (32x16 f16) for output-col block cb from row-major W[(krow0+k)*32 + c]
__device__ __forceinline__ v16h make_bfrag(const float* __restrict__ W, int Kdim, int cb,
                                           int lane, int krow0) {
    const int c  = (lane & 15) + 16 * cb;
    const int kb = (lane >> 4) * 16;
    v16h b;
    #pragma unroll
    for (int t = 0; t < 16; ++t) {
        int k = kb + t;
        float v = (k < Kdim) ? W[(krow0 + k) * EE + c] : 0.0f;
        b[t] = (_Float16)v;
    }
    return b;
}

template <typename T>
__device__ __forceinline__ unsigned lds_off(T* p) {
    // low 32 bits of a generic LDS pointer = LDS byte offset
    return (unsigned)(uintptr_t)p;
}

// Tensor Data Mover: DMA n dwords from global -> LDS (1-row tile, data_size=4B).
// D# bit layout per CDNA5 ISA ch.8 (group0: count/lds_addr/global_addr/type=2;
// group1: data_size, tensor_dim0/1, tile_dim0/1, strides).
__device__ __forceinline__ void tdm_load_dwords(unsigned lds_byte_off, const void* gsrc,
                                                unsigned n) {
    unsigned long long ga = (unsigned long long)(uintptr_t)gsrc;
    v4u g0;
    g0.x = 1u;                                                // count=1, user mode
    g0.y = lds_byte_off;                                      // lds_addr [63:32]
    g0.z = (unsigned)ga;                                      // global_addr [95:64]
    g0.w = (unsigned)((ga >> 32) & 0x01FFFFFFu) | (2u << 30); // addr[56:32] | type=2
    v8i g1;
    g1[0] = (int)(2u << 16);                       // workgroup_mask=0, data_size=4B
    g1[1] = (int)((n & 0xFFFFu) << 16);            // tensor_dim0 lo16 @ bits 63:48
    g1[2] = (int)(((n >> 16) & 0xFFFFu) | (1u << 16)); // tensor_dim0 hi | tensor_dim1=1
    g1[3] = (int)((n & 0xFFFFu) << 16);            // tensor_dim1 hi=0 | tile_dim0=n
    g1[4] = 1;                                     // tile_dim1=1, tile_dim2=0
    g1[5] = (int)n;                                // tensor_dim0_stride lo32
    g1[6] = (int)((n & 0xFFFFu) << 16);            // stride0 hi=0 | stride1 lo16
    g1[7] = (int)((n >> 16) & 0xFFFFu);            // stride1 hi
    v4i z4 = (v4i)0;
#if __has_include(<hip/amd_detail/amd_gfx1250_TDM.h>)
    v8i z8 = (v8i)0;
    __builtin_amdgcn_tensor_load_to_lds(g0, g1, z4, z4, z8, 0);
#else
    __builtin_amdgcn_tensor_load_to_lds(g0, g1, z4, z4, 0);
#endif
}

// ---------------------------------------------------------------------------
// Kernel 0: pre-pack the 10 constant WMMA B-fragments (f16, per-lane v16h) once.
// Frag order: W1[c0,c1], W2[c0,c1], Wo1[c0,c1], Wsx[c0,c1], Wsy[c0,c1]
// ---------------------------------------------------------------------------
__global__ __launch_bounds__(32)
void ltn_pack_weights(const float* __restrict__ W1, const float* __restrict__ W2,
                      const float* __restrict__ Wo1, const float* __restrict__ Ws1,
                      _Float16* __restrict__ packB)
{
    const int lane = threadIdx.x;
    v16h* out = (v16h*)packB;
    out[0 * 32 + lane] = make_bfrag(W1,  FEAT, 0, lane, 0);
    out[1 * 32 + lane] = make_bfrag(W1,  FEAT, 1, lane, 0);
    out[2 * 32 + lane] = make_bfrag(W2,  EE,   0, lane, 0);
    out[3 * 32 + lane] = make_bfrag(W2,  EE,   1, lane, 0);
    out[4 * 32 + lane] = make_bfrag(Wo1, EE,   0, lane, 0);
    out[5 * 32 + lane] = make_bfrag(Wo1, EE,   1, lane, 0);
    out[6 * 32 + lane] = make_bfrag(Ws1, EE,   0, lane, 0);   // Wsx = Ws1[0:32]
    out[7 * 32 + lane] = make_bfrag(Ws1, EE,   1, lane, 0);
    out[8 * 32 + lane] = make_bfrag(Ws1, EE,   0, lane, 32);  // Wsy = Ws1[32:64]
    out[9 * 32 + lane] = make_bfrag(Ws1, EE,   1, lane, 32);
}

// ---------------------------------------------------------------------------
// Kernel 1: fused per-batch kernel. 8 waves = 8 WMMA row-tiles cover the 128
// objects of one batch; encoder results (hx, hy+bs1, iso) land directly in LDS,
// then the O(N^2*E) relu-dot core and per-batch reductions run in-place.
// Inputs are DMA'd into LDS by the Tensor Data Mover (wave 0).
// ---------------------------------------------------------------------------
__global__ __launch_bounds__(256)
void ltn_fused_kernel(const float* __restrict__ boxes, const int* __restrict__ classes,
                      const float* __restrict__ scores, const int* __restrict__ num_objects,
                      const _Float16* __restrict__ packB,
                      const float* __restrict__ b1,  const float* __restrict__ b2,
                      const float* __restrict__ bo1, const float* __restrict__ Wo2,
                      const float* __restrict__ bo2, const float* __restrict__ bs1,
                      const float* __restrict__ Ws2, const float* __restrict__ bs2,
                      float* __restrict__ sat)
{
    __shared__ float s_hx[NN][EE];        // 16 KB
    __shared__ float s_hy[NN][EE];        // 16 KB (bs1 folded in)
    __shared__ float s_stg[8][16][EE];    // 16 KB per-wave D->A re-layout staging
    __shared__ float s_iso[NN];
    __shared__ float s_red[NN];
    __shared__ float s_w2[EE];
    __shared__ float s_box[NN * 4];
    __shared__ float s_sc[NN];
    __shared__ int   s_cls[NN];

    const int b    = blockIdx.x;
    const int tid  = threadIdx.x;
    const int wave = tid >> 5;
    const int lane = tid & 31;
    const int half = lane >> 4;
    const int lm   = lane & 15;
    const int nobj = num_objects[b];

    // ---- TDM: DMA this batch's inputs into LDS (wave 0 issues, EXEC-independent)
    if (tid < 32) {
        tdm_load_dwords(lds_off(&s_box[0]), boxes   + (size_t)b * NN * 4, NN * 4);
        tdm_load_dwords(lds_off(&s_sc[0]),  scores  + (size_t)b * NN,     NN);
        tdm_load_dwords(lds_off(&s_cls[0]), classes + (size_t)b * NN,     NN);
        __builtin_amdgcn_s_wait_tensorcnt(0);
    }
    if (tid < EE) s_w2[tid] = Ws2[tid];
    __syncthreads();

    // ================= encoder: one 16-row tile per wave, 10 WMMAs ==========
    float (*stg)[EE] = s_stg[wave];
    const v16h* frag = (const v16h*)packB;
    const int lrow = wave * 16 + lm;     // this lane's A-matrix row within batch

    // A-frag from features (K = 6 padded to 32; half=1 lanes carry K=16..31 -> zero)
    v16h a;
    #pragma unroll
    for (int t = 0; t < 16; ++t) a[t] = (_Float16)0.0f;
    if (half == 0) {
        a[0] = (_Float16)s_box[lrow * 4 + 0];
        a[1] = (_Float16)s_box[lrow * 4 + 1];
        a[2] = (_Float16)s_box[lrow * 4 + 2];
        a[3] = (_Float16)s_box[lrow * 4 + 3];
        a[4] = (_Float16)(float)s_cls[lrow];
        a[5] = (_Float16)s_sc[lrow];
    }

    // GEMM1: H1 = relu(X @ W1 + b1)
    {
        v8f c0 = wmma_f16(a, frag[0 * 32 + lane]);
        v8f c1 = wmma_f16(a, frag[1 * 32 + lane]);
        #pragma unroll
        for (int v = 0; v < 8; ++v) {
            int m = v + 8 * half;
            stg[m][lm]      = fmaxf(c0[v] + b1[lm],      0.0f);
            stg[m][lm + 16] = fmaxf(c1[v] + b1[lm + 16], 0.0f);
        }
    }
    __syncthreads();
    #pragma unroll
    for (int t = 0; t < 16; ++t) a[t] = (_Float16)stg[lm][a_k(t, half)];
    __syncthreads();

    // GEMM2: enc = H1 @ W2 + b2
    v16h aenc;
    {
        v8f c0 = wmma_f16(a, frag[2 * 32 + lane]);
        v8f c1 = wmma_f16(a, frag[3 * 32 + lane]);
        #pragma unroll
        for (int v = 0; v < 8; ++v) {
            int m = v + 8 * half;
            stg[m][lm]      = c0[v] + b2[lm];
            stg[m][lm + 16] = c1[v] + b2[lm + 16];
        }
    }
    __syncthreads();
    #pragma unroll
    for (int t = 0; t < 16; ++t) aenc[t] = (_Float16)stg[lm][a_k(t, half)];
    __syncthreads();

    // isobj head: G = relu(enc @ Wo1 + bo1); iso = sigmoid(G @ Wo2 + bo2)
    {
        v8f c0 = wmma_f16(aenc, frag[4 * 32 + lane]);
        v8f c1 = wmma_f16(aenc, frag[5 * 32 + lane]);
        #pragma unroll
        for (int v = 0; v < 8; ++v) {
            int m = v + 8 * half;
            stg[m][lm]      = fmaxf(c0[v] + bo1[lm],      0.0f);
            stg[m][lm + 16] = fmaxf(c1[v] + bo1[lm + 16], 0.0f);
        }
    }
    __syncthreads();
    {
        float s = 0.0f;
        #pragma unroll
        for (int n = 0; n < EE; ++n) s += stg[lm][n] * Wo2[n];
        float val = sigmoidf_(s + bo2[0]);
        if (half == 0) s_iso[lrow] = val;   // lanes 0..15 cover the tile's rows
    }

    // hx = enc @ Wsx : D-frags straight into LDS
    {
        v8f c0 = wmma_f16(aenc, frag[6 * 32 + lane]);
        v8f c1 = wmma_f16(aenc, frag[7 * 32 + lane]);
        #pragma unroll
        for (int v = 0; v < 8; ++v) {
            int m = wave * 16 + v + 8 * half;
            s_hx[m][lm]      = c0[v];
            s_hx[m][lm + 16] = c1[v];
        }
    }
    // hy = enc @ Wsy + bs1 (bias folded here)
    {
        v8f c0 = wmma_f16(aenc, frag[8 * 32 + lane]);
        v8f c1 = wmma_f16(aenc, frag[9 * 32 + lane]);
        #pragma unroll
        for (int v = 0; v < 8; ++v) {
            int m = wave * 16 + v + 8 * half;
            s_hy[m][lm]      = c0[v] + bs1[lm];
            s_hy[m][lm + 16] = c1[v] + bs1[lm + 16];
        }
    }
    __syncthreads();

    // ================= pairwise relu-dot core + reductions ==================
    const int i   = tid >> 1;            // 2 threads per row i
    const int sub = tid & 1;
    const float bs2v = bs2[0];

    float hxi[EE];
    #pragma unroll
    for (int e = 0; e < EE; ++e) hxi[e] = s_hx[i][e];

    float acc = 0.0f;
    for (int j = sub; j < nobj; j += 2) {
        float s = 0.0f;
        #pragma unroll
        for (int e = 0; e < EE; ++e)
            s = fmaf(fmaxf(hxi[e] + s_hy[j][e], 0.0f), s_w2[e], s);
        acc += fminf(s_iso[j], sigmoidf_(s + bs2v));
    }

    if (sub == 1) s_red[i] = acc;
    __syncthreads();
    if (sub == 0) {
        float exists = (acc + s_red[i]) / fmaxf((float)nobj, 1.0f);
        float outer  = fmaxf(1.0f - s_iso[i], exists);
        s_red[i] = (i < nobj) ? outer : 1.0f;   // masked rows -> 1.0 per reference
    }
    __syncthreads();
    for (int off = NN / 2; off > 0; off >>= 1) {
        if (tid < off) s_red[tid] = fminf(s_red[tid], s_red[tid + off]);
        __syncthreads();
    }
    if (tid == 0) sat[b] = s_red[0];
}

// ---------------------------------------------------------------------------
// Kernel 2: final masked mean over the 128 batches -> scalar.
// ---------------------------------------------------------------------------
__global__ __launch_bounds__(128)
void ltn_final_kernel(const float* __restrict__ sat, const int* __restrict__ num_objects,
                      float* __restrict__ out)
{
    __shared__ float s_sum[BB];
    __shared__ float s_cnt[BB];
    const int tid = threadIdx.x;
    float v = (num_objects[tid] > 0) ? 1.0f : 0.0f;
    s_sum[tid] = sat[tid] * v;
    s_cnt[tid] = v;
    __syncthreads();
    for (int off = BB / 2; off > 0; off >>= 1) {
        if (tid < off) { s_sum[tid] += s_sum[tid + off]; s_cnt[tid] += s_cnt[tid + off]; }
        __syncthreads();
    }
    if (tid == 0) {
        float cnt = s_cnt[0];
        out[0] = (cnt > 0.0f) ? (s_sum[0] / fmaxf(cnt, 1.0f)) : 1.0f;
    }
}

// ---------------------------------------------------------------------------
extern "C" void kernel_launch(void* const* d_in, const int* in_sizes, int n_in,
                              void* d_out, int out_size, void* d_ws, size_t ws_size,
                              hipStream_t stream) {
    (void)in_sizes; (void)n_in; (void)out_size; (void)ws_size;

    const float* boxes   = (const float*)d_in[0];
    const int*   classes = (const int*)  d_in[1];
    const float* scores  = (const float*)d_in[2];
    const int*   nobj    = (const int*)  d_in[3];
    const float* W1      = (const float*)d_in[4];
    const float* b1      = (const float*)d_in[5];
    const float* W2      = (const float*)d_in[6];
    const float* b2      = (const float*)d_in[7];
    const float* Wo1     = (const float*)d_in[8];
    const float* bo1     = (const float*)d_in[9];
    const float* Wo2     = (const float*)d_in[10];
    const float* bo2     = (const float*)d_in[11];
    const float* Ws1     = (const float*)d_in[12];
    const float* bs1     = (const float*)d_in[13];
    const float* Ws2     = (const float*)d_in[14];
    const float* bs2     = (const float*)d_in[15];

    float*     ws    = (float*)d_ws;
    float*     sat   = ws;                               // BB floats
    _Float16*  packB = (_Float16*)(ws + BB);             // 10*32*16 halves (10 KB)

    ltn_pack_weights<<<1, 32, 0, stream>>>(W1, W2, Wo1, Ws1, packB);

    ltn_fused_kernel<<<BB, 256, 0, stream>>>(
        boxes, classes, scores, nobj, packB,
        b1, b2, bo1, Wo2, bo2, bs1, Ws2, bs2, sat);

    ltn_final_kernel<<<1, BB, 0, stream>>>(sat, nobj, (float*)d_out);
}
